// Encoder_71021579207308
// MI455X (gfx1250) — compile-verified, matrix operand order checked by reading, not run on hardware
//
#include <hip/hip_runtime.h>
#include <hip/hip_bf16.h>
#include <math.h>

// ---------------------------------------------------------------------------
// CDNA5 (gfx1250) wave32 WMMA bf16 path.
// ---------------------------------------------------------------------------
typedef __attribute__((ext_vector_type(16))) __bf16 v16bf;
typedef __attribute__((ext_vector_type(8)))  float  v8f;
typedef int v4i __attribute__((vector_size(16)));

#define E_DIM 256
#define HEADS 8
#define HD 32
#define FF_DIM 1024
#define NB 8            // batch
#define TP 1056         // padded tokens (mult of 32; >= 1028 keys rounded to 32)
#define RTOT (NB * TP)  // 8448 rows
#define LDSP 36         // LDS row pitch (floats): 16B aligned, conflict-spread

__device__ __forceinline__ v8f wmma_bf16(v16bf a, v16bf b, v8f c) {
  return __builtin_amdgcn_wmma_f32_16x16x32_bf16(
      /*neg_a=*/false, a, /*neg_b=*/false, b,
      /*c_mod=*/(short)0, c, /*reuse_a=*/false, /*reuse_b=*/false);
}

// ---- async global->LDS copy (16B per lane), with safe fallback -------------
#if __has_builtin(__builtin_amdgcn_global_load_async_to_lds_b128)
typedef __attribute__((address_space(1))) v4i* as1_v4i_ptr;
typedef __attribute__((address_space(3))) v4i* as3_v4i_ptr;
__device__ __forceinline__ void async_cp16(void* lds_dst, const void* glb_src) {
  __builtin_amdgcn_global_load_async_to_lds_b128(
      (as1_v4i_ptr)(unsigned long long)glb_src,
      (as3_v4i_ptr)(unsigned int)(unsigned long long)lds_dst,
      0, 0);
}
template <int N>
__device__ __forceinline__ void async_wait_le() {
#if __has_builtin(__builtin_amdgcn_s_wait_asynccnt)
  __builtin_amdgcn_s_wait_asynccnt(N);
#else
  if constexpr (N == 0)      asm volatile("s_wait_asynccnt 0" ::: "memory");
  else if constexpr (N == 6) asm volatile("s_wait_asynccnt 6" ::: "memory");
  else                       asm volatile("s_wait_asynccnt 8" ::: "memory");
#endif
}
#else
__device__ __forceinline__ void async_cp16(void* lds_dst, const void* glb_src) {
  *(float4*)lds_dst = *(const float4*)glb_src;
}
template <int N>
__device__ __forceinline__ void async_wait_le() {}
#endif

// A-fragment loader: 16x32 bf16 tile from row-major f32, optional scale.
// lane&15 = row; low lanes hold k in [0,8)+[16,24), high lanes [8,16)+[24,32).
__device__ __forceinline__ v16bf load_frag_rowmajor(const float* __restrict__ base,
                                                    int ld, int row0, int kbase,
                                                    float scale) {
  int lane = threadIdx.x & 31;
  int r15  = lane & 15;
  int kh   = (lane < 16) ? 0 : 8;
  const float* p = base + (size_t)(row0 + r15) * ld + kbase;
  v16bf a;
#pragma unroll
  for (int j = 0; j < 8; ++j) a[j]     = (__bf16)(p[kh + j] * scale);
#pragma unroll
  for (int j = 0; j < 8; ++j) a[8 + j] = (__bf16)(p[kh + 16 + j] * scale);
  return a;
}

__device__ __forceinline__ float rowmax16(float x) {
#pragma unroll
  for (int m = 1; m < 16; m <<= 1) x = fmaxf(x, __shfl_xor(x, m, 16));
  return x;
}
__device__ __forceinline__ float rowsum16(float x) {
#pragma unroll
  for (int m = 1; m < 16; m <<= 1) x += __shfl_xor(x, m, 16);
  return x;
}

// ---------------------------------------------------------------------------
// Generic GEMM: Y[r][c] = act( sum_k X[r][k] * W[c][k] + bias[c] )
// Block = 128 thr (4 waves): 64-row x (NCT*16)-col tile, K-steps of 32.
// Double-buffered LDS staging via async b128 copies: tile k+1 is in flight
// while tile k feeds the WMMA pipes. NCT compile-time so accumulators stay
// in fixed VGPRs (no v_movrel).
// ---------------------------------------------------------------------------
template <int NCT>
__global__ __launch_bounds__(128) void gemm_xwT(
    const float* __restrict__ X, int ldx, long xz,
    const float* __restrict__ W, int ldw, long wz,
    float* __restrict__ Y, int ldy, long yz,
    const float* __restrict__ bias,
    int K, int relu) {
  __shared__ float Xs[2][64][LDSP];
  __shared__ float Ws[2][NCT * 16][LDSP];

  X += (long)blockIdx.z * xz;
  W += (long)blockIdx.z * wz;
  Y += (long)blockIdx.z * yz;

  const int tid  = threadIdx.x;
  const int lane = tid & 31;
  const int wave = tid >> 5;
  const int r15  = lane & 15;
  const int kh   = (lane < 16) ? 0 : 8;
  const int row0 = blockIdx.x * 64;          // block row base
  const int col0 = blockIdx.y * (NCT * 16);  // block col base

  auto stage = [&](int kb, int buf) {
#pragma unroll
    for (int i = 0; i < 4; ++i) {
      int f = tid + i * 128;     // float4 index, 0..511
      int row = f >> 3, kq = (f & 7) * 4;
      async_cp16(&Xs[buf][row][kq], X + (size_t)(row0 + row) * ldx + kb + kq);
    }
#pragma unroll
    for (int i = 0; i < NCT; ++i) {
      int f = tid + i * 128;
      int row = f >> 3, kq = (f & 7) * 4;
      async_cp16(&Ws[buf][row][kq], W + (size_t)(col0 + row) * ldw + kb + kq);
    }
  };

  v8f zero = {};
  v8f acc[NCT];
#pragma unroll
  for (int c = 0; c < NCT; ++c) acc[c] = zero;

  stage(0, 0);
  int cur = 0;
  for (int kb = 0; kb < K; kb += 32) {
    const bool more = (kb + 32) < K;
    if (more) {
      stage(kb + 32, cur ^ 1);
      async_wait_le<4 + NCT>();  // prior group (tile kb) complete; next in flight
    } else {
      async_wait_le<0>();
    }
    __syncthreads();

    // ---- fragments from LDS, WMMA --------------------------------------
    v16bf a;
    {
      const float* px = &Xs[cur][wave * 16 + r15][0];
#pragma unroll
      for (int j = 0; j < 8; ++j) a[j]     = (__bf16)px[kh + j];
#pragma unroll
      for (int j = 0; j < 8; ++j) a[8 + j] = (__bf16)px[kh + 16 + j];
    }
#pragma unroll
    for (int ct = 0; ct < NCT; ++ct) {
      v16bf b;
      const float* pw = &Ws[cur][ct * 16 + r15][0];
#pragma unroll
      for (int j = 0; j < 8; ++j) b[j]     = (__bf16)pw[kh + j];
#pragma unroll
      for (int j = 0; j < 8; ++j) b[8 + j] = (__bf16)pw[kh + 16 + j];
      acc[ct] = wmma_bf16(a, b, acc[ct]);
    }
    __syncthreads();   // done reading buf `cur`; safe to overwrite later
    cur ^= 1;
  }

  const int rowoff = (lane < 16) ? 0 : 8;
  const int wrow0  = row0 + wave * 16;
#pragma unroll
  for (int ct = 0; ct < NCT; ++ct) {
    int col  = col0 + ct * 16 + r15;
    float bv = bias ? bias[col] : 0.0f;
#pragma unroll
    for (int r = 0; r < 8; ++r) {
      float val = acc[ct][r] + bv;
      if (relu) val = fmaxf(val, 0.0f);
      Y[(size_t)(wrow0 + r + rowoff) * ldy + col] = val;
    }
  }
}

// ---------------------------------------------------------------------------
// Flash attention: one wave per 16-query tile, per (n, h).
// q/k/v/out: f32 [N, TP, E], head h occupies cols [h*32, h*32+32).
// Online softmax over 32-key chunks. Q pre-scaled by 1/sqrt(E)=1/16.
// ---------------------------------------------------------------------------
__global__ __launch_bounds__(128) void attention_kernel(
    const float* __restrict__ qg, const float* __restrict__ kg,
    const float* __restrict__ vg, float* __restrict__ outg, int Tvalid) {
  __shared__ __bf16 Pl[4][16][32];

  int lane = threadIdx.x & 31;
  int wave = threadIdx.x >> 5;
  int nh = blockIdx.y;
  int n = nh >> 3, h = nh & 7;
  int qtile = blockIdx.x * 4 + wave;
  if (qtile >= (TP >> 4)) return;

  const size_t base = (size_t)n * TP * E_DIM + h * HD;
  const float* qp = qg + base;
  const float* kp = kg + base;
  const float* vp = vg + base;
  float*       op = outg + base;

  int r15 = lane & 15;
  int kh  = (lane < 16) ? 0 : 8;
  int rowoff = (lane < 16) ? 0 : 8;

  v16bf qf = load_frag_rowmajor(qp, E_DIM, qtile * 16, 0, 0.0625f);

  float mrow[8], lrow[8];
  v8f o0 = {}, o1 = {};
#pragma unroll
  for (int r = 0; r < 8; ++r) { mrow[r] = -1e30f; lrow[r] = 0.0f; }

  for (int kb = 0; kb < TP; kb += 32) {
    // ---- energy: two 16x16 halves, E = Q * K^T (K rows load like A frags)
    v8f e0, e1;
    {
      v16bf kf0 = load_frag_rowmajor(kp, E_DIM, kb, 0, 1.0f);
      v16bf kf1 = load_frag_rowmajor(kp, E_DIM, kb + 16, 0, 1.0f);
      v8f z = {};
      e0 = wmma_bf16(qf, kf0, z);
      e1 = wmma_bf16(qf, kf1, z);
    }
    bool ok0 = (kb + r15) < Tvalid;
    bool ok1 = (kb + 16 + r15) < Tvalid;

    float p0[8], p1[8];
#pragma unroll
    for (int r = 0; r < 8; ++r) {
      float x0 = ok0 ? e0[r] : -1e30f;
      float x1 = ok1 ? e1[r] : -1e30f;
      float rm = rowmax16(fmaxf(x0, x1));
      float mnew = fmaxf(mrow[r], rm);
      float alpha = __expf(mrow[r] - mnew);
      float q0 = ok0 ? __expf(x0 - mnew) : 0.0f;
      float q1 = ok1 ? __expf(x1 - mnew) : 0.0f;
      lrow[r] = lrow[r] * alpha + rowsum16(q0 + q1);
      o0[r] *= alpha;
      o1[r] *= alpha;
      mrow[r] = mnew;
      p0[r] = q0;
      p1[r] = q1;
    }

    // ---- C-layout P -> A-layout via per-wave LDS slab (in-wave, no barrier)
#pragma unroll
    for (int r = 0; r < 8; ++r) {
      Pl[wave][r + rowoff][r15]      = (__bf16)p0[r];
      Pl[wave][r + rowoff][16 + r15] = (__bf16)p1[r];
    }
    v16bf pf;
#pragma unroll
    for (int j = 0; j < 8; ++j) pf[j]     = Pl[wave][r15][kh + j];
#pragma unroll
    for (int j = 0; j < 8; ++j) pf[8 + j] = Pl[wave][r15][kh + 16 + j];

    // ---- O += P * V  (B image = column-gather of V; zero masked keys)
#pragma unroll
    for (int dh = 0; dh < 2; ++dh) {
      const float* vcol = vp + dh * 16 + r15;
      v16bf vf;
#pragma unroll
      for (int j = 0; j < 8; ++j) {
        int key = kb + kh + j;
        vf[j] = (__bf16)((key < Tvalid) ? vcol[(size_t)key * E_DIM] : 0.0f);
      }
#pragma unroll
      for (int j = 0; j < 8; ++j) {
        int key = kb + kh + 16 + j;
        vf[8 + j] = (__bf16)((key < Tvalid) ? vcol[(size_t)key * E_DIM] : 0.0f);
      }
      if (dh == 0) o0 = wmma_bf16(pf, vf, o0);
      else         o1 = wmma_bf16(pf, vf, o1);
    }
  }

#pragma unroll
  for (int r = 0; r < 8; ++r) {
    float inv = 1.0f / lrow[r];
    float* dst = op + (size_t)(qtile * 16 + r + rowoff) * E_DIM;
    dst[r15]      = o0[r] * inv;
    dst[16 + r15] = o1[r] * inv;
  }
}

// ---------------------------------------------------------------------------
// dst[row] = LN(a[row] + b[row]) * g + beta   (width 256, one block per row)
// ---------------------------------------------------------------------------
__global__ __launch_bounds__(256) void ln_residual(
    float* __restrict__ dst, const float* __restrict__ a,
    const float* __restrict__ b, const float* __restrict__ g,
    const float* __restrict__ beta) {
  __shared__ float red[256];
  size_t row = blockIdx.x;
  int e = threadIdx.x;
  float x = a[row * E_DIM + e] + b[row * E_DIM + e];
  red[e] = x;
  __syncthreads();
  for (int s = 128; s > 0; s >>= 1) { if (e < s) red[e] += red[e + s]; __syncthreads(); }
  float mu = red[0] * (1.0f / 256.0f);
  __syncthreads();
  float d = x - mu;
  red[e] = d * d;
  __syncthreads();
  for (int s = 128; s > 0; s >>= 1) { if (e < s) red[e] += red[e + s]; __syncthreads(); }
  float var = red[0] * (1.0f / 256.0f);
  dst[row * E_DIM + e] = d * rsqrtf(var + 1e-5f) * g[e] + beta[e];
}

// Final: out[n,s,:] = LN_512( concat(x0[n,s,:], x1[n,s,:]) )
__global__ __launch_bounds__(512) void final_ln(
    float* __restrict__ out, const float* __restrict__ x0,
    const float* __restrict__ x1, const float* __restrict__ g,
    const float* __restrict__ beta) {
  __shared__ float red[512];
  int t = blockIdx.x;            // 0 .. N*1024-1
  int n = t >> 10, s = t & 1023;
  int e = threadIdx.x;           // 0 .. 511
  size_t roff = ((size_t)n * TP + s) * E_DIM;
  float x = (e < 256) ? x0[roff + e] : x1[roff + (e - 256)];
  red[e] = x;
  __syncthreads();
  for (int st = 256; st > 0; st >>= 1) { if (e < st) red[e] += red[e + st]; __syncthreads(); }
  float mu = red[0] * (1.0f / 512.0f);
  __syncthreads();
  float d = x - mu;
  red[e] = d * d;
  __syncthreads();
  for (int st = 256; st > 0; st >>= 1) { if (e < st) red[e] += red[e + st]; __syncthreads(); }
  float var = red[0] * (1.0f / 512.0f);
  out[(size_t)t * 512 + e] = d * rsqrtf(var + 1e-5f) * g[e] + beta[e];
}

// x_ws[n, s, e] = x_in[n, s, e] + pe(s, e) for s < 1024 ; 0 for pad rows
__global__ void pos_add(float* __restrict__ dst, const float* __restrict__ src) {
  size_t idx = (size_t)blockIdx.x * blockDim.x + threadIdx.x;
  size_t total = (size_t)NB * TP * E_DIM;
  if (idx >= total) return;
  int e = (int)(idx % E_DIM);
  size_t nt = idx / E_DIM;
  int s = (int)(nt % TP);
  int n = (int)(nt / TP);
  float v = 0.0f;
  if (s < 1024) {
    int j = e >> 1;
    float ang = (float)s * powf(10000.0f, -(float)(4 * j) / 256.0f);
    float pe = (e & 1) ? cosf(ang) : sinf(ang);
    v = src[((size_t)n * 1024 + s) * E_DIM + e] + pe;
  }
  dst[idx] = v;
}

// rows 1024..1027 <- bn ; rows 1028..TP-1 <- 0
__global__ void set_bn(float* __restrict__ x, const float* __restrict__ bn) {
  size_t idx = (size_t)blockIdx.x * blockDim.x + threadIdx.x;
  const int PADR = TP - 1024;
  size_t total = (size_t)NB * PADR * E_DIM;
  if (idx >= total) return;
  int e = (int)(idx % E_DIM);
  size_t nj = idx / E_DIM;
  int j = (int)(nj % PADR);
  int n = (int)(nj / PADR);
  float v = (j < 4) ? bn[((size_t)n * 4 + j) * E_DIM + e] : 0.0f;
  x[((size_t)n * TP + 1024 + j) * E_DIM + e] = v;
}

__global__ void bn_mean(float* __restrict__ bn, const float* __restrict__ x0,
                        const float* __restrict__ x1) {
  int idx = blockIdx.x * blockDim.x + threadIdx.x;
  if (idx >= NB * 4 * E_DIM) return;
  int e = idx % E_DIM;
  int nj = idx / E_DIM;
  int j = nj % 4, n = nj / 4;
  size_t off = ((size_t)n * TP + 1024 + j) * E_DIM + e;
  bn[idx] = 0.5f * (x0[off] + x1[off]);
}

__global__ void copyk(float* __restrict__ d, const float* __restrict__ s, int ncnt) {
  int i = blockIdx.x * blockDim.x + threadIdx.x;
  if (i < ncnt) d[i] = s[i];
}

// ---------------------------------------------------------------------------
extern "C" void kernel_launch(void* const* d_in, const int* in_sizes, int n_in,
                              void* d_out, int out_size, void* d_ws, size_t ws_size,
                              hipStream_t stream) {
  (void)in_sizes; (void)n_in; (void)out_size; (void)ws_size;
  const float* x_a   = (const float*)d_in[0];
  const float* x_b   = (const float*)d_in[1];
  const float* bneck = (const float*)d_in[2];
  const float* Wv    = (const float*)d_in[3];
  const float* Wk    = (const float*)d_in[4];
  const float* Wq    = (const float*)d_in[5];
  const float* Wo    = (const float*)d_in[6];
  const float* bo    = (const float*)d_in[7];
  const float* ln1g  = (const float*)d_in[8];
  const float* ln1b  = (const float*)d_in[9];
  const float* ln2g  = (const float*)d_in[10];
  const float* ln2b  = (const float*)d_in[11];
  const float* Wff1  = (const float*)d_in[12];
  const float* bff1  = (const float*)d_in[13];
  const float* Wff2  = (const float*)d_in[14];
  const float* bff2  = (const float*)d_in[15];
  const float* outg  = (const float*)d_in[16];
  const float* outb  = (const float*)d_in[17];

  const size_t RE = (size_t)RTOT * E_DIM;   // 8448 * 256
  float* ws = (float*)d_ws;
  float* xbuf[2] = { ws, ws + RE };
  float* x1buf = ws + 2 * RE;
  float* attnb = ws + 3 * RE;
  float* ybuf  = ws + 4 * RE;
  float* qb    = ws + 5 * RE;
  float* kbuf  = ws + 6 * RE;
  float* vb    = ws + 7 * RE;
  float* h1    = ws + 8 * RE;
  float* bnb   = h1 + (size_t)RTOT * FF_DIM;

  // ---- init: positional encoding + zero pad rows; bn <- input bottleneck
  {
    size_t tot = (size_t)NB * TP * E_DIM;
    pos_add<<<(unsigned)((tot + 255) / 256), 256, 0, stream>>>(xbuf[0], x_a);
    pos_add<<<(unsigned)((tot + 255) / 256), 256, 0, stream>>>(xbuf[1], x_b);
    copyk<<<(NB * 4 * E_DIM + 255) / 256, 256, 0, stream>>>(bnb, bneck, NB * 4 * E_DIM);
  }

  for (int lyr = 0; lyr < 4; ++lyr) {
    const bool fusion = (lyr >= 2);
    const int Tvalid = fusion ? 1028 : 1024;
    if (fusion) {
      size_t tot = (size_t)NB * (TP - 1024) * E_DIM;
      set_bn<<<(unsigned)((tot + 255) / 256), 256, 0, stream>>>(xbuf[0], bnb);
      set_bn<<<(unsigned)((tot + 255) / 256), 256, 0, stream>>>(xbuf[1], bnb);
    }
    for (int m = 0; m < 2; ++m) {
      const size_t pidx = (size_t)lyr * 2 + m;
      // --- QKV projections (per-head z-batched; weight shared across heads)
      dim3 gq(RTOT / 64, 1, HEADS);
      gemm_xwT<2><<<gq, 128, 0, stream>>>(xbuf[m], E_DIM, 32L, Wq + pidx * HD * HD, HD, 0L,
                                          qb,   E_DIM, 32L, nullptr, HD, 0);
      gemm_xwT<2><<<gq, 128, 0, stream>>>(xbuf[m], E_DIM, 32L, Wk + pidx * HD * HD, HD, 0L,
                                          kbuf, E_DIM, 32L, nullptr, HD, 0);
      gemm_xwT<2><<<gq, 128, 0, stream>>>(xbuf[m], E_DIM, 32L, Wv + pidx * HD * HD, HD, 0L,
                                          vb,   E_DIM, 32L, nullptr, HD, 0);
      // --- attention (flash, per 16-query tile per head)
      dim3 ga((TP / 16 + 3) / 4, NB * HEADS);
      attention_kernel<<<ga, 128, 0, stream>>>(qb, kbuf, vb, attnb, Tvalid);
      // --- output projection + bias
      dim3 go(RTOT / 64, E_DIM / 64);
      gemm_xwT<4><<<go, 128, 0, stream>>>(attnb, E_DIM, 0L, Wo + pidx * E_DIM * E_DIM, E_DIM, 0L,
                                          ybuf, E_DIM, 0L, bo + pidx * E_DIM, E_DIM, 0);
      // --- x1 = LN(attn_out + x)
      ln_residual<<<RTOT, 256, 0, stream>>>(x1buf, ybuf, xbuf[m],
                                            ln1g + pidx * E_DIM, ln1b + pidx * E_DIM);
      // --- FFN
      dim3 g1(RTOT / 64, FF_DIM / 64);
      gemm_xwT<4><<<g1, 128, 0, stream>>>(x1buf, E_DIM, 0L, Wff1 + pidx * FF_DIM * E_DIM, E_DIM, 0L,
                                          h1, FF_DIM, 0L, bff1 + pidx * FF_DIM, E_DIM, 1);
      dim3 g2(RTOT / 64, E_DIM / 64);
      gemm_xwT<4><<<g2, 128, 0, stream>>>(h1, FF_DIM, 0L, Wff2 + pidx * E_DIM * FF_DIM, FF_DIM, 0L,
                                          ybuf, E_DIM, 0L, bff2 + pidx * E_DIM, FF_DIM, 0);
      // --- x = LN(ff_out + x1)
      ln_residual<<<RTOT, 256, 0, stream>>>(xbuf[m], ybuf, x1buf,
                                            ln2g + pidx * E_DIM, ln2b + pidx * E_DIM);
    }
    if (fusion)
      bn_mean<<<(NB * 4 * E_DIM + 255) / 256, 256, 0, stream>>>(bnb, xbuf[0], xbuf[1]);
  }

  // ---- out = LN_512(concat(x0, x1))
  final_ln<<<NB * 1024, 512, 0, stream>>>((float*)d_out, xbuf[0], xbuf[1], outg, outb);
}
